// MetricLoss_37546604102166
// MI455X (gfx1250) — matrix operand order repaired and checked by reading, not compiled, not run on hardware
//
#include <hip/hip_runtime.h>

// ---- types for WMMA fragments (wave32, gfx1250) ----
typedef __attribute__((ext_vector_type(16))) _Float16 v16h;
typedef __attribute__((ext_vector_type(8)))  _Float16 v8h;
typedef __attribute__((ext_vector_type(8)))  float    v8f;

#define EMB_D 128          // embedding dim (K)
#define MARGIN_F 1.0f

// ---------------------------------------------------------------------------
// f32 -> f16 pre-conversion of both embedding matrices (done once; saves
// per-tile cvt work and halves bandwidth in the GEMM passes).
// ---------------------------------------------------------------------------
__global__ __launch_bounds__(256) void cvt_f32_to_f16_kernel(
    const float* __restrict__ a, const float* __restrict__ b,
    _Float16* __restrict__ ah, _Float16* __restrict__ bh, int total) {
  int i = blockIdx.x * blockDim.x + threadIdx.x;
  if (i < total) {
    ah[i] = (_Float16)a[i];
    bh[i] = (_Float16)b[i];
  }
}

// Load A-fragment (16x32 f16): lane = 16*h + m; runs [k0+8h,+8) and [k0+16+8h,+8)
__device__ __forceinline__ v16h load_a_frag(const _Float16* __restrict__ row,
                                            int k0, int h) {
  v16h f;
  *((v8h*)&f + 0) = *(const v8h*)(row + k0 + h * 8);
  *((v8h*)&f + 1) = *(const v8h*)(row + k0 + 16 + h * 8);
  return f;
}

// Load B-fragment (32x16 f16): lane = 16*h + n; runs [k0+8h,+8) and [k0+16+8h,+8) of col n
__device__ __forceinline__ v16h load_b_frag(const _Float16* __restrict__ row,
                                            int k0, int h) {
  v16h f;
  *((v8h*)&f + 0) = *(const v8h*)(row + k0 + h * 8);
  *((v8h*)&f + 1) = *(const v8h*)(row + k0 + 16 + h * 8);
  return f;
}

// ---------------------------------------------------------------------------
// Pass 1:  V[i] = sum_{j : label[j] != label[i]} exp(margin + S_ij)
// One block (8 waves) per 16-row tile; each wave sweeps 1/8 of the column tiles.
// ---------------------------------------------------------------------------
__global__ __launch_bounds__(256) void v_pass_kernel(
    const _Float16* __restrict__ ah, const _Float16* __restrict__ bh,
    const int* __restrict__ labels, float* __restrict__ V, int N) {
  const int lane = threadIdx.x & 31;
  const int wave = threadIdx.x >> 5;   // 0..7
  const int h    = lane >> 4;          // half-wave id
  const int ln   = lane & 15;
  const int iBase = blockIdx.x * 16;
  const int numJ  = N >> 4;

  // A fragments for all 4 K-steps (row = iBase + ln), reused for every j tile
  const _Float16* arow = ah + (size_t)(iBase + ln) * EMB_D;
  v16h afrag[4];
#pragma unroll
  for (int kk = 0; kk < 4; ++kk) afrag[kk] = load_a_frag(arow, kk * 32, h);

  int labi[8];
#pragma unroll
  for (int r = 0; r < 8; ++r) labi[r] = labels[iBase + h * 8 + r];

  float vpart[8];
#pragma unroll
  for (int r = 0; r < 8; ++r) vpart[r] = 0.0f;

  for (int jt = wave; jt < numJ; jt += 8) {
    const int jBase = jt * 16;
    const _Float16* brow = bh + (size_t)(jBase + ln) * EMB_D;
    v8f acc = {};
#pragma unroll
    for (int kk = 0; kk < 4; ++kk) {
      v16h bfrag = load_b_frag(brow, kk * 32, h);
      acc = __builtin_amdgcn_wmma_f32_16x16x32_f16(
          false, afrag[kk], false, bfrag, (short)0, acc, false, false);
    }
    const int labj = labels[jBase + ln];
#pragma unroll
    for (int r = 0; r < 8; ++r) {
      if (labi[r] != labj) vpart[r] += __expf(MARGIN_F + acc[r]);
    }
  }

  // Row m = 8h + r lives across the 16 lanes of this half-wave: reduce there.
#pragma unroll
  for (int r = 0; r < 8; ++r) {
    float v = vpart[r];
    v += __shfl_xor(v, 8, 32);
    v += __shfl_xor(v, 4, 32);
    v += __shfl_xor(v, 2, 32);
    v += __shfl_xor(v, 1, 32);
    if (ln == 0) atomicAdd(&V[iBase + h * 8 + r], v);
  }
}

// ---------------------------------------------------------------------------
// Pass 2: sum over positive pairs (same label, i != j) of
//         max(log(V_i + V_j) - S_ij, 0)^2, plus positive-pair count.
// Positive pairs are ~0.1% dense: screen each 16x16 tile with a wave-uniform
// label match test and skip the WMMAs/loads/logs when no pair is present.
// The branch condition is wave-uniform, so EXEC is all-1s at every WMMA.
// ---------------------------------------------------------------------------
__global__ __launch_bounds__(256) void loss_pass_kernel(
    const _Float16* __restrict__ ah, const _Float16* __restrict__ bh,
    const int* __restrict__ labels, const float* __restrict__ V,
    float* __restrict__ loss_sum, unsigned int* __restrict__ pos_cnt, int N) {
  const int lane = threadIdx.x & 31;
  const int wave = threadIdx.x >> 5;
  const int h    = lane >> 4;
  const int ln   = lane & 15;
  const int iBase = blockIdx.x * 16;
  const int numJ  = N >> 4;

  const _Float16* arow = ah + (size_t)(iBase + ln) * EMB_D;
  v16h afrag[4];
#pragma unroll
  for (int kk = 0; kk < 4; ++kk) afrag[kk] = load_a_frag(arow, kk * 32, h);

  int   labi[8];
  float Vi[8];
#pragma unroll
  for (int r = 0; r < 8; ++r) {
    labi[r] = labels[iBase + h * 8 + r];
    Vi[r]   = V[iBase + h * 8 + r];
  }

  float        lpart = 0.0f;
  unsigned int cpart = 0u;

  for (int jt = wave; jt < numJ; jt += 8) {
    const int jBase = jt * 16;
    const int j     = jBase + ln;
    const int labj  = labels[j];

    // Tile screen: does any (row label, column label) pair match?
    // Each lane checks its column's label against all 8 of its row labels;
    // between the two half-waves this covers the full 16x16 label product.
    int match = 0;
#pragma unroll
    for (int r = 0; r < 8; ++r) match |= (labi[r] == labj);

    if (__any(match)) {   // wave-uniform branch: EXEC stays all-1s inside
      const _Float16* brow = bh + (size_t)j * EMB_D;
      v8f acc = {};
#pragma unroll
      for (int kk = 0; kk < 4; ++kk) {
        v16h bfrag = load_b_frag(brow, kk * 32, h);
        acc = __builtin_amdgcn_wmma_f32_16x16x32_f16(
            false, afrag[kk], false, bfrag, (short)0, acc, false, false);
      }
      const float Vj = V[j];
#pragma unroll
      for (int r = 0; r < 8; ++r) {
        const int i = iBase + h * 8 + r;
        if (labi[r] == labj && i != j) {
          float hinge = fmaxf(__logf(Vi[r] + Vj) - acc[r], 0.0f);
          lpart += hinge * hinge;
          cpart += 1u;
        }
      }
    }
  }

  // Full-wave reduction (flat scalar sums)
  lpart += __shfl_xor(lpart, 16, 32);
  lpart += __shfl_xor(lpart, 8, 32);
  lpart += __shfl_xor(lpart, 4, 32);
  lpart += __shfl_xor(lpart, 2, 32);
  lpart += __shfl_xor(lpart, 1, 32);
  cpart += __shfl_xor(cpart, 16, 32);
  cpart += __shfl_xor(cpart, 8, 32);
  cpart += __shfl_xor(cpart, 4, 32);
  cpart += __shfl_xor(cpart, 2, 32);
  cpart += __shfl_xor(cpart, 1, 32);
  if (lane == 0) {
    atomicAdd(loss_sum, lpart);
    atomicAdd(pos_cnt, cpart);
  }
}

__global__ void finalize_kernel(const float* __restrict__ loss_sum,
                                const unsigned int* __restrict__ pos_cnt,
                                float* __restrict__ out) {
  if (threadIdx.x == 0 && blockIdx.x == 0) {
    out[0] = loss_sum[0] / (2.0f * (float)pos_cnt[0]);
  }
}

// ---------------------------------------------------------------------------
extern "C" void kernel_launch(void* const* d_in, const int* in_sizes, int n_in,
                              void* d_out, int out_size, void* d_ws, size_t ws_size,
                              hipStream_t stream) {
  const float* a      = (const float*)d_in[0];
  const float* b      = (const float*)d_in[1];
  const int*   labels = (const int*)d_in[2];
  float*       out    = (float*)d_out;

  const int N = in_sizes[2];         // 8192
  const int total = N * EMB_D;       // elements per matrix

  // workspace layout: [ V : N floats | loss_sum : f32 | pos_cnt : u32 | pad | ah | bh ]
  float*        V        = (float*)d_ws;
  float*        loss_sum = V + N;
  unsigned int* pos_cnt  = (unsigned int*)(V + N + 1);
  size_t        half_off = (((size_t)(N + 8) * sizeof(float)) + 255) & ~(size_t)255;
  _Float16*     ah       = (_Float16*)((char*)d_ws + half_off);
  _Float16*     bh       = ah + (size_t)total;

  // zero V + accumulators (graph-capturable async memset)
  hipMemsetAsync(d_ws, 0, (size_t)(N + 8) * sizeof(float), stream);

  // f32 -> f16 conversion of both matrices
  {
    int threads = 256;
    int blocks  = (total + threads - 1) / threads;
    cvt_f32_to_f16_kernel<<<blocks, threads, 0, stream>>>(a, b, ah, bh, total);
  }

  const int rowTiles = N / 16;  // 512
  v_pass_kernel<<<rowTiles, 256, 0, stream>>>(ah, bh, labels, V, N);
  loss_pass_kernel<<<rowTiles, 256, 0, stream>>>(ah, bh, labels, V, loss_sum, pos_cnt, N);
  finalize_kernel<<<1, 32, 0, stream>>>(loss_sum, pos_cnt, out);
}
